// CrossAttentionFusion_26551487824139
// MI455X (gfx1250) — matrix operand order repaired and checked by reading, not compiled, not run on hardware
//
#include <hip/hip_runtime.h>
#include <hip/hip_bf16.h>
#include <math.h>

#define NTOK 4096   // H*W
#define CCH  128    // channels
#define NB   4      // batch

typedef __attribute__((ext_vector_type(16))) __bf16 v16bf;
typedef __attribute__((ext_vector_type(8)))  __bf16 v8bf;
typedef __attribute__((ext_vector_type(8)))  float  v8f;

static __device__ __forceinline__ v16bf cat8(v8bf lo, v8bf hi) {
  return __builtin_shufflevector(lo, hi, 0,1,2,3,4,5,6,7,8,9,10,11,12,13,14,15);
}
static __device__ __forceinline__ v8bf cvt8(float4 a, float4 b) {
  v8bf r;
  r[0] = (__bf16)a.x; r[1] = (__bf16)a.y; r[2] = (__bf16)a.z; r[3] = (__bf16)a.w;
  r[4] = (__bf16)b.x; r[5] = (__bf16)b.y; r[6] = (__bf16)b.z; r[7] = (__bf16)b.w;
  return r;
}
static __device__ __forceinline__ v8f wmma_bf16(v16bf a, v16bf b, v8f c) {
  // D = A(16x32) x B(32x16) + C, f32 accumulate
  return __builtin_amdgcn_wmma_f32_16x16x32_bf16(false, a, false, b, (short)0, c, false, false);
}

// ---------------------------------------------------------------------------
// Transpose [B][C][N] f32 -> [B][N][C] bf16  (32x32 LDS tile transpose)
// ---------------------------------------------------------------------------
__global__ __launch_bounds__(256)
void transpose_to_bf16(const float* __restrict__ in, __bf16* __restrict__ out) {
  __shared__ float tile[32][33];
  const int n0 = blockIdx.x * 32;
  const int c0 = blockIdx.y * 32;
  const int b  = blockIdx.z;
  const int tx = threadIdx.x, ty = threadIdx.y;
  const float* src = in  + (size_t)b * CCH * NTOK;
  __bf16*      dst = out + (size_t)b * NTOK * CCH;
  #pragma unroll
  for (int i = 0; i < 4; ++i)
    tile[ty + 8*i][tx] = src[(size_t)(c0 + ty + 8*i) * NTOK + n0 + tx];
  __syncthreads();
  #pragma unroll
  for (int i = 0; i < 4; ++i)
    dst[(size_t)(n0 + ty + 8*i) * CCH + c0 + tx] = (__bf16)tile[tx][ty + 8*i];
}

// ---------------------------------------------------------------------------
// 1x1 conv as WMMA GEMM: Y[b,o,n] = sum_c W[o,c]*X[b,c,n] (+ W2*X2) + bias[o]
// XT is [B][N][C] bf16 (B-matrix tiles load per-lane contiguous in c).
// MODE 0: store bf16 [N][C] (transposed)        -> for Q, K
// MODE 1: store bf16 [C][N]                     -> for V
// MODE 2: store f32  [N][C], sigmoid applied    -> gate
// MODE 3: store f32  [C][N]                     -> final output
// DUAL: add W2 @ X2T (for the channel-concat gate conv), branch-free.
// Block 256 (8 waves, each owns 16 output rows), grid (N/64, B).
// ---------------------------------------------------------------------------
template <int MODE, bool DUAL>
__global__ __launch_bounds__(256)
void conv1x1_wmma(const float* __restrict__ W, int ldw,
                  const __bf16* __restrict__ XT,
                  const float* __restrict__ W2,
                  const __bf16* __restrict__ X2T,
                  const float* __restrict__ bias,
                  void* __restrict__ outp) {
  const int lane = threadIdx.x & 31;
  const int wid  = threadIdx.x >> 5;
  const int hi   = lane >> 4;
  const int lo16 = lane & 15;
  const int b    = blockIdx.y;
  const int n0   = blockIdx.x * 64;
  const int o0   = wid * 16;

  const __bf16* xt  = XT + (size_t)b * NTOK * CCH;
  const __bf16* x2t = DUAL ? (X2T + (size_t)b * NTOK * CCH) : nullptr;

  v8f acc[4] = {};

  #pragma unroll
  for (int cs = 0; cs < 4; ++cs) {
    const int c0 = cs * 32;
    // A tile: W rows, bf16-converted; per-lane chunk [16hi,16hi+16)
    const float* wr = W + (size_t)(o0 + lo16) * ldw + c0 + hi * 16;
    v16bf a = cat8(cvt8(*(const float4*)(wr),      *(const float4*)(wr + 4)),
                   cvt8(*(const float4*)(wr + 8), *(const float4*)(wr + 12)));
    v16bf a2;
    if (DUAL) {
      const float* w2r = W2 + (size_t)(o0 + lo16) * ldw + c0 + hi * 16;
      a2 = cat8(cvt8(*(const float4*)(w2r),      *(const float4*)(w2r + 4)),
                cvt8(*(const float4*)(w2r + 8), *(const float4*)(w2r + 12)));
    }
    #pragma unroll
    for (int t = 0; t < 4; ++t) {
      const int n = n0 + t * 16 + lo16;
      v16bf bm = *(const v16bf*)(xt + (size_t)n * CCH + c0 + hi * 16);
      acc[t] = wmma_bf16(a, bm, acc[t]);
      if (DUAL) {
        v16bf b2 = *(const v16bf*)(x2t + (size_t)n * CCH + c0 + hi * 16);
        acc[t] = wmma_bf16(a2, b2, acc[t]);
      }
    }
  }

  float bv[8];
  #pragma unroll
  for (int j = 0; j < 8; ++j) bv[j] = bias[o0 + hi * 8 + j];

  if (MODE == 0) {              // bf16 [N][C] transposed
    __bf16* ob = (__bf16*)outp + (size_t)b * NTOK * CCH;
    #pragma unroll
    for (int t = 0; t < 4; ++t) {
      const int n = n0 + t * 16 + lo16;
      v8bf pk;
      #pragma unroll
      for (int j = 0; j < 8; ++j) pk[j] = (__bf16)(acc[t][j] + bv[j]);
      *(v8bf*)(ob + (size_t)n * CCH + o0 + hi * 8) = pk;
    }
  } else if (MODE == 1) {       // bf16 [C][N]
    __bf16* ob = (__bf16*)outp + (size_t)b * CCH * NTOK;
    #pragma unroll
    for (int t = 0; t < 4; ++t) {
      const int n = n0 + t * 16 + lo16;
      #pragma unroll
      for (int j = 0; j < 8; ++j)
        ob[(size_t)(o0 + hi * 8 + j) * NTOK + n] = (__bf16)(acc[t][j] + bv[j]);
    }
  } else if (MODE == 2) {       // f32 [N][C] + sigmoid
    float* of = (float*)outp + (size_t)b * NTOK * CCH;
    #pragma unroll
    for (int t = 0; t < 4; ++t) {
      const int n = n0 + t * 16 + lo16;
      #pragma unroll
      for (int j = 0; j < 8; ++j) {
        float v = acc[t][j] + bv[j];
        of[(size_t)n * CCH + o0 + hi * 8 + j] = 1.0f / (1.0f + __expf(-v));
      }
    }
  } else {                      // f32 [C][N]
    float* of = (float*)outp + (size_t)b * CCH * NTOK;
    #pragma unroll
    for (int t = 0; t < 4; ++t) {
      const int n = n0 + t * 16 + lo16;
      #pragma unroll
      for (int j = 0; j < 8; ++j)
        of[(size_t)(o0 + hi * 8 + j) * NTOK + n] = acc[t][j] + bv[j];
    }
  }
}

// ---------------------------------------------------------------------------
// Flash attention (channels-first): out[n][c] = softmax_m(q_n.k_m/sqrt(C)).v
// Qt,Kt: [B][N][C] bf16;  V: [B][C][N] bf16;  outT: [B][N][C] f32
// Block 256 (8 waves x 16 queries), grid (N/128, B). Key tiles of 64:
// 32 WMMAs per tile vs one softmax pass -> low VALU/ds overhead per FLOP.
// K/V B-tiles loaded straight from global (L2-resident: 1MB/batch each),
// with global_prefetch of the next tile overlapping current-tile WMMAs.
// LDS only used to relayout P from D-layout to A-layout (2KB per wave).
// ---------------------------------------------------------------------------
__global__ __launch_bounds__(256)
void flash_attn_wmma(const __bf16* __restrict__ Qt, const __bf16* __restrict__ Kt,
                     const __bf16* __restrict__ V, float* __restrict__ outT,
                     float scale) {
  const int lane = threadIdx.x & 31;
  const int wid  = threadIdx.x >> 5;
  const int hi   = lane >> 4;
  const int lo16 = lane & 15;
  const int b    = blockIdx.y;
  const int q0   = blockIdx.x * 128 + wid * 16;

  const __bf16* qt = Qt + (size_t)b * NTOK * CCH;
  const __bf16* kt = Kt + (size_t)b * NTOK * CCH;
  const __bf16* vv = V  + (size_t)b * CCH * NTOK;
  float*        ot = outT + (size_t)b * NTOK * CCH;

  __shared__ __attribute__((aligned(32))) __bf16 plds[8][16][64];

  // Q A-tiles resident for the whole key loop (4 c-steps of 32)
  v16bf aq[4];
  {
    const __bf16* qr = qt + (size_t)(q0 + lo16) * CCH + hi * 16;
    #pragma unroll
    for (int cs = 0; cs < 4; ++cs)
      aq[cs] = cat8(*(const v8bf*)(qr + cs * 32), *(const v8bf*)(qr + cs * 32 + 8));
  }

  v8f   o_acc[8] = {};
  float m_i[8], l_i[8];
  #pragma unroll
  for (int j = 0; j < 8; ++j) { m_i[j] = -3.0e38f; l_i[j] = 0.0f; }

  for (int m0 = 0; m0 < NTOK; m0 += 64) {
    // prefetch next key tile's K rows and V rows into cache (speculative;
    // dropped silently when past the end of the tensor)
    {
      const int mn = m0 + 64;
      #pragma unroll
      for (int t = 0; t < 4; ++t)
        __builtin_prefetch(kt + (size_t)(mn + t * 16 + lo16) * CCH + hi * 16, 0, 1);
      __builtin_prefetch(vv + (size_t)(wid * 16 + lo16) * NTOK + mn + hi * 16, 0, 1);
    }

    // ---- S = scale * Qt.K  (four 16-key tiles, k over 128 channels) ----
    v8f s[4] = {};
    #pragma unroll
    for (int t = 0; t < 4; ++t) {
      const __bf16* kr = kt + (size_t)(m0 + t * 16 + lo16) * CCH + hi * 16;
      #pragma unroll
      for (int cs = 0; cs < 4; ++cs) {
        v16bf bk = *(const v16bf*)(kr + cs * 32);
        s[t] = wmma_bf16(aq[cs], bk, s[t]);
      }
    }
    #pragma unroll
    for (int t = 0; t < 4; ++t)
      #pragma unroll
      for (int j = 0; j < 8; ++j) s[t][j] *= scale;

    // ---- online softmax: row = query (j + 8*hi), cols across 16 lanes ----
    float esc[8];
    #pragma unroll
    for (int j = 0; j < 8; ++j) {
      float mx = fmaxf(fmaxf(s[0][j], s[1][j]), fmaxf(s[2][j], s[3][j]));
      #pragma unroll
      for (int off = 8; off >= 1; off >>= 1) mx = fmaxf(mx, __shfl_xor(mx, off, 16));
      float mnew = fmaxf(m_i[j], mx);
      esc[j] = __expf(m_i[j] - mnew);
      m_i[j] = mnew;
    }
    #pragma unroll
    for (int t = 0; t < 4; ++t)
      #pragma unroll
      for (int j = 0; j < 8; ++j) s[t][j] = __expf(s[t][j] - m_i[j]);
    #pragma unroll
    for (int j = 0; j < 8; ++j) {
      float r = (s[0][j] + s[1][j]) + (s[2][j] + s[3][j]);
      #pragma unroll
      for (int off = 8; off >= 1; off >>= 1) r += __shfl_xor(r, off, 16);
      l_i[j] = l_i[j] * esc[j] + r;
    }
    #pragma unroll
    for (int ct = 0; ct < 8; ++ct)
      #pragma unroll
      for (int j = 0; j < 8; ++j) o_acc[ct][j] *= esc[j];

    // ---- relayout P: D-layout -> LDS -> A-layout (wave-private tile) ----
    #pragma unroll
    for (int t = 0; t < 4; ++t)
      #pragma unroll
      for (int j = 0; j < 8; ++j)
        plds[wid][j + 8 * hi][t * 16 + lo16] = (__bf16)s[t][j];
    v16bf ap0 = cat8(*(const v8bf*)&plds[wid][lo16][hi * 16],
                     *(const v8bf*)&plds[wid][lo16][8 + hi * 16]);
    v16bf ap1 = cat8(*(const v8bf*)&plds[wid][lo16][32 + hi * 16],
                     *(const v8bf*)&plds[wid][lo16][40 + hi * 16]);

    // ---- O += P.V^T  (8 channel tiles, k over 64 keys) ----
    #pragma unroll
    for (int ct = 0; ct < 8; ++ct) {
      const __bf16* vr = vv + (size_t)(ct * 16 + lo16) * NTOK + m0 + hi * 16;
      v16bf bv0 = *(const v16bf*)(vr);
      v16bf bv1 = *(const v16bf*)(vr + 32);
      o_acc[ct] = wmma_bf16(ap0, bv0, o_acc[ct]);
      o_acc[ct] = wmma_bf16(ap1, bv1, o_acc[ct]);
    }
  }

  // normalize and store [N][C] f32
  float inv[8];
  #pragma unroll
  for (int j = 0; j < 8; ++j) inv[j] = 1.0f / l_i[j];
  #pragma unroll
  for (int ct = 0; ct < 8; ++ct)
    #pragma unroll
    for (int j = 0; j < 8; ++j)
      ot[(size_t)(q0 + j + 8 * hi) * CCH + ct * 16 + lo16] = o_acc[ct][j] * inv[j];
}

// ---------------------------------------------------------------------------
// fused = g*out1 + (1-g)*out2  (all [B][N][C]); result bf16 for final GEMM
// ---------------------------------------------------------------------------
__global__ __launch_bounds__(256)
void gate_fuse(const float* __restrict__ g, const float* __restrict__ o1,
               const float* __restrict__ o2, __bf16* __restrict__ fused, int n) {
  for (int i = blockIdx.x * blockDim.x + threadIdx.x; i < n; i += gridDim.x * blockDim.x) {
    float gv = g[i];
    fused[i] = (__bf16)(gv * o1[i] + (1.0f - gv) * o2[i]);
  }
}

// ---------------------------------------------------------------------------
extern "C" void kernel_launch(void* const* d_in, const int* in_sizes, int n_in,
                              void* d_out, int out_size, void* d_ws, size_t ws_size,
                              hipStream_t stream) {
  (void)in_sizes; (void)n_in; (void)out_size; (void)ws_size;
  const float* hsi = (const float*)d_in[0];
  const float* msi = (const float*)d_in[1];
  const float* wq1 = (const float*)d_in[2];  const float* bq1 = (const float*)d_in[3];
  const float* wk1 = (const float*)d_in[4];  const float* bk1 = (const float*)d_in[5];
  const float* wv1 = (const float*)d_in[6];  const float* bv1 = (const float*)d_in[7];
  const float* wq2 = (const float*)d_in[8];  const float* bq2 = (const float*)d_in[9];
  const float* wk2 = (const float*)d_in[10]; const float* bk2 = (const float*)d_in[11];
  const float* wv2 = (const float*)d_in[12]; const float* bv2 = (const float*)d_in[13];
  const float* wg  = (const float*)d_in[14]; const float* bg  = (const float*)d_in[15];
  const float* wp  = (const float*)d_in[16]; const float* bp  = (const float*)d_in[17];

  // workspace carve-up (~60 MB total)
  char* p = (char*)d_ws;
  const size_t szBF = (size_t)NB * NTOK * CCH * sizeof(__bf16); // 4 MB
  const size_t szF  = (size_t)NB * NTOK * CCH * sizeof(float);  // 8 MB
  __bf16* hsiT = (__bf16*)p; p += szBF;
  __bf16* msiT = (__bf16*)p; p += szBF;
  __bf16* q1t  = (__bf16*)p; p += szBF;
  __bf16* k1t  = (__bf16*)p; p += szBF;
  __bf16* q2t  = (__bf16*)p; p += szBF;
  __bf16* k2t  = (__bf16*)p; p += szBF;
  __bf16* v1m  = (__bf16*)p; p += szBF;
  __bf16* v2m  = (__bf16*)p; p += szBF;
  float*  gT   = (float*)p;  p += szF;
  float*  o1T  = (float*)p;  p += szF;
  float*  o2T  = (float*)p;  p += szF;
  __bf16* fusT = (__bf16*)p; p += szBF;

  dim3 tb(32, 8), tg(NTOK / 32, CCH / 32, NB);
  transpose_to_bf16<<<tg, tb, 0, stream>>>(hsi, hsiT);
  transpose_to_bf16<<<tg, tb, 0, stream>>>(msi, msiT);

  dim3 cb(256), cg(NTOK / 64, NB);
  conv1x1_wmma<0, false><<<cg, cb, 0, stream>>>(wq1, CCH, hsiT, nullptr, nullptr, bq1, q1t);
  conv1x1_wmma<0, false><<<cg, cb, 0, stream>>>(wk1, CCH, msiT, nullptr, nullptr, bk1, k1t);
  conv1x1_wmma<1, false><<<cg, cb, 0, stream>>>(wv1, CCH, msiT, nullptr, nullptr, bv1, v1m);
  conv1x1_wmma<0, false><<<cg, cb, 0, stream>>>(wq2, CCH, msiT, nullptr, nullptr, bq2, q2t);
  conv1x1_wmma<0, false><<<cg, cb, 0, stream>>>(wk2, CCH, hsiT, nullptr, nullptr, bk2, k2t);
  conv1x1_wmma<1, false><<<cg, cb, 0, stream>>>(wv2, CCH, hsiT, nullptr, nullptr, bv2, v2m);
  // gate: wg is [C][2C]; cols [0,128)=hsi, [128,256)=msi
  conv1x1_wmma<2, true><<<cg, cb, 0, stream>>>(wg, 2 * CCH, hsiT, wg + CCH, msiT, bg, gT);

  const float scale = 1.0f / sqrtf((float)CCH);
  dim3 fg(NTOK / 128, NB);
  flash_attn_wmma<<<fg, cb, 0, stream>>>(q1t, k1t, v1m, o1T, scale);
  flash_attn_wmma<<<fg, cb, 0, stream>>>(q2t, k2t, v2m, o2T, scale);

  gate_fuse<<<2048, 256, 0, stream>>>(gT, o1T, o2T, fusT, NB * NTOK * CCH);

  conv1x1_wmma<3, false><<<cg, cb, 0, stream>>>(wp, CCH, fusT, nullptr, nullptr, bp, d_out);
}
